// BiDenseConv2d_45947560132915
// MI455X (gfx1250) — matrix operand order repaired and checked by reading, not compiled, not run on hardware
//
#include <hip/hip_runtime.h>
#include <math.h>

typedef __attribute__((ext_vector_type(16))) _Float16 v16h;
typedef __attribute__((ext_vector_type(8)))  _Float16 v8h;
typedef __attribute__((ext_vector_type(8)))  float    v8f;
typedef __attribute__((ext_vector_type(4)))  unsigned int u32x4;
typedef __attribute__((ext_vector_type(8)))  int i32x8;
typedef __attribute__((ext_vector_type(4)))  int i32x4;

// ---- problem constants ----
constexpr int BATCH = 8;
constexpr int CIN   = 48;
constexpr int COUT  = 64;
constexpr int HWD   = 224;                 // H == W
constexpr int PHW   = HWD + 2;             // 226: halo-padded spatial dim
constexpr int NPIX  = BATCH * HWD * HWD;   // 401408 output pixels
constexpr int NPPAD = BATCH * PHW * PHW;   // 408608 padded pixels
constexpr int CPAD  = 64;                  // input channels padded for clean K chunks
constexpr int NCHUNK = 18;                 // K = 9 taps * 64 = 576 -> 18 chunks of 32
constexpr int W2TILES = HWD / 32;          // 7 (each wave: 32 pixels = 2 M-subtiles)
constexpr int NWAVE   = NPIX / 32;         // 12544 conv waves
constexpr int K2_BLOCKS = NWAVE / 8;       // 1568 blocks of 8 waves
constexpr int WB_HALFS  = 4 * NCHUNK * 32 * 16;  // 36864 packed weight halfs
constexpr float TWO_PI_F = 6.28318530717958647692f;

// ---- workspace layout (bytes), all 256B aligned ----
constexpr size_t XA_OFF   = 0;
constexpr size_t XA_BYTES = (size_t)NPPAD * CPAD * 2;         // 52,301,824 (halo incl.)
constexpr size_t Y_OFF    = XA_OFF + XA_BYTES;
constexpr size_t Y_BYTES  = (size_t)NPIX * COUT * 2;          // 51,380,224
constexpr size_t WB_OFF   = Y_OFF + Y_BYTES;
constexpr size_t WB_BYTES = (size_t)WB_HALFS * 2;             // 73,728
constexpr size_t SC_OFF   = WB_OFF + WB_BYTES;                // scale[64] f32
constexpr size_t PART_OFF = SC_OFF + 256;
constexpr size_t PART_BYTES = (size_t)K2_BLOCKS * 128 * 4;    // 802,816
constexpr size_t ABN_OFF  = PART_OFF + PART_BYTES;            // a[64], b[64] f32
// total ~105 MB

// ---------------------------------------------------------------------------
// Kernel 0a: pack binarized weights (+/-1 f16, exact) in B-fragment-linear
// layout. B order used here (contiguous k, matches A): element j of v16h -> K = j + 16*(lane>=16),
// N = lane%16.  flat index f = ((n*18 + k)*32 + lane)*16 + j
// ---------------------------------------------------------------------------
__global__ void k_prep_wb(const float* __restrict__ w, _Float16* __restrict__ wB) {
    int f = blockIdx.x * 256 + threadIdx.x;               // < 36864
    int j    = f & 15;
    int lane = (f >> 4) & 31;
    int k    = (f >> 9) % NCHUNK;
    int n    = (f >> 9) / NCHUNK;
    int K    = k * 32 + j + ((lane >> 4) << 4);           // B-operand K mapping
    int o    = n * 16 + (lane & 15);
    int tap  = K >> 6;                                    // kh*3+kw
    int c    = K & 63;
    _Float16 v = (_Float16)0.f;
    if (c < CIN) {
        float wv = w[(o * CIN + c) * 9 + tap];            // OIHW flat: tap = kh*3+kw
        v = (_Float16)((wv > 0.f) ? 1.f : (wv < 0.f ? -1.f : 0.f));
    }
    wB[f] = v;
}

// Kernel 0b: per-filter scale = mean |w| (64 filters x 432 taps)
__global__ void k_scale(const float* __restrict__ w, float* __restrict__ scale) {
    int o = threadIdx.x;                                  // 64 threads
    float s = 0.f;
    for (int i = 0; i < CIN * 9; ++i) s += fabsf(w[o * CIN * 9 + i]);
    scale[o] = s * (1.f / (CIN * 9));
}

// ---------------------------------------------------------------------------
// Kernel 1: binarize activations into a halo-padded channels-last f16 tensor
// xa[b, 1+h, 1+w, c] = sign(sin((x - eps)/tau * 2pi)) * A[c]  (0 for c>=48),
// border pixels (conv zero padding) written as all-zero rows.
// ---------------------------------------------------------------------------
__global__ void k_binx(const float* __restrict__ x, const float* __restrict__ eps,
                       const float* __restrict__ tau, const float* __restrict__ Aa,
                       _Float16* __restrict__ xa) {
    int p = blockIdx.x * 256 + threadIdx.x;               // padded pixel id
    if (p >= NPPAD) return;
    int col = p % PHW;
    int row = (p / PHW) % PHW;
    int b   = p / (PHW * PHW);
    alignas(32) _Float16 vals[CPAD];
    bool interior = (col >= 1) && (col <= HWD) && (row >= 1) && (row <= HWD);
    if (interior) {
        int w = col - 1, h = row - 1;
        for (int c = 0; c < CIN; ++c) {
            float xv = x[((size_t)(b * CIN + c) * HWD + h) * HWD + w];
            float z  = (xv - eps[c]) / tau[c] * TWO_PI_F;
            float s  = sinf(z);
            float sg = (s > 0.f) ? 1.f : ((s < 0.f) ? -1.f : 0.f);
            vals[c] = (_Float16)(sg * Aa[c]);
        }
        #pragma unroll
        for (int c = CIN; c < CPAD; ++c) vals[c] = (_Float16)0.f;
    } else {
        #pragma unroll
        for (int c = 0; c < CPAD; ++c) vals[c] = (_Float16)0.f;
    }
    v8h* dst = (v8h*)(xa + (size_t)p * CPAD);
    const v8h* src = (const v8h*)vals;
    #pragma unroll
    for (int i = 0; i < 8; ++i) dst[i] = src[i];
}

// ---------------------------------------------------------------------------
// Kernel 2: implicit-GEMM 3x3 conv via v_wmma_f32_16x16x32_f16.
// Wave = 32 pixels (two 16-pixel M-subtiles) x 64 out channels (4 N-tiles).
// Weights staged per block into LDS via the Tensor Data Mover and read with
// ds_load_b128.  With the halo-padded activation tensor every A-fragment load
// is an unconditional immediate-offset global_load_b128 off a single per-lane
// base address (no exec masking, no zero fills).
// ---------------------------------------------------------------------------
__global__ void __launch_bounds__(256) k_conv(const _Float16* __restrict__ xa,
                                              const _Float16* __restrict__ wB,
                                              const float* __restrict__ scale,
                                              _Float16* __restrict__ y,
                                              float* __restrict__ partial) {
    __shared__ _Float16 sB[WB_HALFS];       // 73,728 B of packed weight fragments
    __shared__ float lsum[COUT];
    __shared__ float lsq[COUT];
    int tid = threadIdx.x;
    if (tid < 64) lsum[tid] = 0.f;
    else if (tid < 128) lsq[tid - 64] = 0.f;

#if __has_builtin(__builtin_amdgcn_tensor_load_to_lds)
    if (tid == 0) {
        unsigned lds_off = (unsigned)(size_t)(void*)sB;
        unsigned long long ga = (unsigned long long)(size_t)wB;
        u32x4 g0;
        g0.x = 1u;                                        // count=1 (valid user D#)
        g0.y = lds_off;                                   // lds_addr
        g0.z = (unsigned)(ga & 0xffffffffu);              // global_addr[31:0]
        g0.w = (unsigned)((ga >> 32) & 0x01ffffffu)       // global_addr[56:32]
             | (2u << 30);                                // type = 2 ("image")
        i32x8 g1;
        g1[0] = (1 << 16);                                // data_size=2B, mask=0
        g1[1] = (int)((WB_HALFS & 0xffff) << 16);         // tensor_dim0[15:0]
        g1[2] = (int)((WB_HALFS >> 16) & 0xffff);         // tensor_dim0[31:16]
        g1[2] |= (1 << 16);                               // tensor_dim1 = 1
        g1[3] = (int)((WB_HALFS & 0xffff) << 16);         // tile_dim0 = 36864
        g1[4] = 1;                                        // tile_dim1 = 1
        g1[5] = WB_HALFS;                                 // tensor_dim0_stride
        g1[6] = 0; g1[7] = 0;
        i32x4 gz = {0, 0, 0, 0};                          // groups 2/3 unused (<=2D)
#if defined(__clang_major__) && (__clang_major__ >= 23)
        i32x8 gz8 = {0, 0, 0, 0, 0, 0, 0, 0};
        __builtin_amdgcn_tensor_load_to_lds(g0, g1, gz, gz, gz8, 0);
#else
        __builtin_amdgcn_tensor_load_to_lds(g0, g1, gz, gz, 0);
#endif
        __builtin_amdgcn_s_wait_tensorcnt(0);
    }
#else
    {
        const v8h* gsrc = (const v8h*)wB;
        v8h* ldst = (v8h*)sB;
        for (int i = tid; i < WB_HALFS / 8; i += 256) ldst[i] = gsrc[i];
    }
#endif
    __syncthreads();

    int lane = tid & 31;
    int s    = blockIdx.x * 8 + (tid >> 5);   // wave id, grid exact
    int wc   = s % W2TILES;
    int t2   = s / W2TILES;
    int h    = t2 % HWD;
    int b    = t2 / HWD;
    int w0   = wc * 32;
    int m    = lane & 15;                     // pixel within subtile
    int hi   = lane >> 4;

    // single per-lane base: padded pixel (row=h, col=w0+m) of image b
    const _Float16* base = xa + ((size_t)((b * PHW + h) * PHW + w0 + m)) * CPAD
                              + hi * 16;

    v8f acc[8] = {};                          // [0..3]: subtile0, [4..7]: subtile1
    const v16h* sf = (const v16h*)sB;

    #pragma unroll
    for (int k = 0; k < NCHUNK; ++k) {
        const int tap   = k >> 1;
        const int kh    = tap / 3;
        const int kw    = tap % 3;
        const int chalf = (k & 1) * 32;
        // compile-time immediate offsets after full unroll
        const int off = (kh * PHW + kw) * CPAD + chalf;
        union { v16h v; v8h h2[2]; } ua0, ua1;
        ua0.h2[0] = *(const v8h*)(base + off);
        ua0.h2[1] = *(const v8h*)(base + off + 8);
        ua1.h2[0] = *(const v8h*)(base + off + 16 * CPAD);
        ua1.h2[1] = *(const v8h*)(base + off + 16 * CPAD + 8);
        #pragma unroll
        for (int n = 0; n < 4; ++n) {
            v16h bf = sf[(n * NCHUNK + k) * 32 + lane];   // ds_load_b128 x2
            acc[n]     = __builtin_amdgcn_wmma_f32_16x16x32_f16(
                false, ua0.v, false, bf, (short)0, acc[n],     false, false);
            acc[4 + n] = __builtin_amdgcn_wmma_f32_16x16x32_f16(
                false, ua1.v, false, bf, (short)0, acc[4 + n], false, false);
        }
    }

    #pragma unroll
    for (int n = 0; n < 4; ++n) {
        int o = n * 16 + m;                   // D element: N = lane%16, M = r + hi*8
        float sc = scale[o];
        float ss = 0.f, qq = 0.f;
        #pragma unroll
        for (int r = 0; r < 8; ++r) {
            float v0 = acc[n][r] * sc;        // exact f32 per-filter scaling pre-BN
            float v1 = acc[4 + n][r] * sc;
            acc[n][r] = v0; acc[4 + n][r] = v1;
            ss += v0 + v1; qq += v0 * v0 + v1 * v1;
        }
        ss += __shfl_xor(ss, 16, 32);         // combine wave halves (same o)
        qq += __shfl_xor(qq, 16, 32);
        if (hi == 0) { atomicAdd(&lsum[o], ss); atomicAdd(&lsq[o], qq); }
        #pragma unroll
        for (int r = 0; r < 8; ++r) {
            int mm = r + hi * 8;
            size_t obase = (size_t)((b * HWD + h) * HWD + w0 + mm);
            y[obase * COUT + o]        = (_Float16)acc[n][r];
            y[(obase + 16) * COUT + o] = (_Float16)acc[4 + n][r];
        }
    }
    __syncthreads();
    if (tid < 64)       partial[(size_t)blockIdx.x * 128 + tid] = lsum[tid];
    else if (tid < 128) partial[(size_t)blockIdx.x * 128 + tid] = lsq[tid - 64];
}

// ---------------------------------------------------------------------------
// Kernel 3: reduce per-block partials -> BN affine  a[o]=gamma*rstd,
// b[o]=beta - mu*a  (128 threads, contention-free strided reduction)
// ---------------------------------------------------------------------------
__global__ void k_stats(const float* __restrict__ partial, const float* __restrict__ gamma,
                        const float* __restrict__ beta, float* __restrict__ abn) {
    __shared__ float red[128];
    int t = threadIdx.x;                      // 128
    float s = 0.f;
    for (int i = 0; i < K2_BLOCKS; ++i) s += partial[(size_t)i * 128 + t];
    red[t] = s;
    __syncthreads();
    if (t < COUT) {
        float cnt  = (float)NPIX;
        float mu   = red[t] / cnt;
        float var  = red[64 + t] / cnt - mu * mu;   // biased variance
        float rstd = rsqrtf(var + 1e-5f);
        float a    = gamma[t] * rstd;
        abn[t]      = a;
        abn[64 + t] = beta[t] - mu * a;
    }
}

// ---------------------------------------------------------------------------
// Kernel 4: out = y*a[o] + b[o] + channel_adaptive_bypass(x)
// thread = one (b,h,w); y row is 128B contiguous; x/out coalesced over w.
// ---------------------------------------------------------------------------
__global__ void k_final(const float* __restrict__ x, const _Float16* __restrict__ y,
                        const float* __restrict__ abn, float* __restrict__ out) {
    __shared__ float sa[COUT], sb[COUT];
    int t = threadIdx.x;
    if (t < 64) sa[t] = abn[t];
    else if (t < 128) sb[t - 64] = abn[t];
    __syncthreads();

    int p = blockIdx.x * 256 + t;             // pixel id, grid exact
    int w = p % HWD;
    int h = (p / HWD) % HWD;
    int b = p / (HWD * HWD);

    float xv[CIN];
    for (int c = 0; c < CIN; ++c)
        xv[c] = x[((size_t)(b * CIN + c) * HWD + h) * HWD + w];

    alignas(32) _Float16 yl[COUT];
    const v8h* ysrc = (const v8h*)(y + (size_t)p * COUT);
    v8h* yd = (v8h*)yl;
    #pragma unroll
    for (int i = 0; i < 8; ++i) yd[i] = ysrc[i];

    float* op = out + (size_t)b * COUT * HWD * HWD + (size_t)h * HWD + w;
    #pragma unroll
    for (int o = 0; o < COUT; ++o) {
        float bp;
        if (o < CIN) bp = xv[o];
        else if (o < 63) { int j = o - 48; bp = (xv[j] + xv[j + 15] + xv[j + 30]) * (1.f / 3.f); }
        else bp = (xv[45] + xv[46] + xv[47]) * (1.f / 3.f);
        float yv = (float)yl[o];
        op[(size_t)o * HWD * HWD] = yv * sa[o] + sb[o] + bp;
    }
}

// ---------------------------------------------------------------------------
extern "C" void kernel_launch(void* const* d_in, const int* in_sizes, int n_in,
                              void* d_out, int out_size, void* d_ws, size_t ws_size,
                              hipStream_t stream) {
    const float* x     = (const float*)d_in[0];
    // d_in[1] = alpha: forward value is sign(); alpha only shapes the STE gradient
    const float* eps   = (const float*)d_in[2];
    const float* tau   = (const float*)d_in[3];
    const float* Aa    = (const float*)d_in[4];
    const float* wgt   = (const float*)d_in[5];
    const float* gamma = (const float*)d_in[6];
    const float* beta  = (const float*)d_in[7];
    float* out = (float*)d_out;

    char* ws = (char*)d_ws;
    _Float16* xa    = (_Float16*)(ws + XA_OFF);
    _Float16* yb    = (_Float16*)(ws + Y_OFF);
    _Float16* wB    = (_Float16*)(ws + WB_OFF);
    float* scale    = (float*)(ws + SC_OFF);
    float* partial  = (float*)(ws + PART_OFF);
    float* abn      = (float*)(ws + ABN_OFF);

    k_prep_wb<<<144, 256, 0, stream>>>(wgt, wB);
    k_scale <<<1, 64, 0, stream>>>(wgt, scale);
    k_binx  <<<(NPPAD + 255) / 256, 256, 0, stream>>>(x, eps, tau, Aa, xa);
    k_conv  <<<K2_BLOCKS, 256, 0, stream>>>(xa, wB, scale, yb, partial);
    k_stats <<<1, 128, 0, stream>>>(partial, gamma, beta, abn);
    k_final <<<NPIX / 256, 256, 0, stream>>>(x, yb, abn, out);
}
